// SingleAttentionHead_9363028705719
// MI455X (gfx1250) — matrix operand order repaired and checked by reading, not compiled, not run on hardware
//
#include <hip/hip_runtime.h>
#include <hip/hip_bf16.h>

// ---- problem dims ----
#define B_ 8
#define S_ 2048
#define D_ 1024
#define H_ 64

typedef __attribute__((ext_vector_type(16))) __bf16 v16bf;
typedef __attribute__((ext_vector_type(8)))  float  v8f;

__device__ __forceinline__ __bf16 f2bf(float f) { return (__bf16)f; }

// A-matrix 16x32 bf16 layout: lane m = lane&15, hi = lane>>4.
// VGPR v holds K pair {2v + 16*hi, +1} (contiguous, matches B).
__device__ __forceinline__ int a_k_of(int v, int hi) {
  return 2 * v + 16 * hi;
}
// B-matrix 32x16 bf16 layout: lane n = lane&15, hi = lane>>4.
// VGPR v holds K pair {2v + 16*hi, +1} -> per-lane contiguous K range [16*hi, 16*hi+16).
__device__ __forceinline__ int b_k_of(int v, int hi) { return 2 * v + 16 * hi; }

// LDS offset of a __shared__ object (flat addr low 32 bits == LDS offset).
__device__ __forceinline__ unsigned lds_off(const void* p) {
  return (unsigned)(unsigned long long)p;
}

// ---------------------------------------------------------------------------
// Kernel 0: prep — transpose+convert W into WT[3][H][D] bf16 so that the
// WMMA B-operand per lane becomes a contiguous 32-byte read.
// Grid (H, 3), block 256.
// ---------------------------------------------------------------------------
__global__ __launch_bounds__(256) void prep_kernel(
    const float* __restrict__ Wq, const float* __restrict__ Wk,
    const float* __restrict__ Wv, __bf16* __restrict__ WT) {
  const int h = blockIdx.x;   // 0..H-1
  const int m = blockIdx.y;   // 0..2
  const float* W = (m == 0) ? Wq : (m == 1) ? Wk : Wv;
  __bf16* dst = WT + ((size_t)m * H_ + h) * D_;
  for (int d = threadIdx.x; d < D_; d += 256)
    dst[d] = f2bf(W[(size_t)d * H_ + h]);
}

// ---------------------------------------------------------------------------
// Kernel 1: QKV projection.  Grid (S/16, B), block = 96 threads (3 waves).
// Wave w computes matrix w (q/k/v): 16 rows x 64 cols, K over D in 32-steps.
// x tile (16x32 f32, 2 KB) staged in LDS via global_load_async_to_lds_b128
// issued by wave 0 (ASYNCcnt path), converted to bf16 on LDS read.
// v is written TRANSPOSED: vT[B][H][S].
// ---------------------------------------------------------------------------
__global__ __launch_bounds__(96) void proj_kernel(
    const float* __restrict__ x, const __bf16* __restrict__ WT,
    __bf16* __restrict__ q, __bf16* __restrict__ k, __bf16* __restrict__ vT) {
  __shared__ float xt[16][32];   // raw f32 tile

  const int row0 = blockIdx.x * 16;
  const int b    = blockIdx.y;
  const int tid  = threadIdx.x;
  const int wave = tid >> 5;     // 0..2 -> q,k,v
  const int lane = tid & 31;
  const int m  = lane & 15;
  const int hi = lane >> 4;
  const int n  = lane & 15;

  const __bf16* Wrow = WT + (size_t)wave * H_ * D_;   // [H][D], row h contiguous

  v8f acc[4] = {};
  const float* xrow = x + ((size_t)b * S_ + row0) * D_;
  const unsigned xt_base = lds_off(&xt[0][0]);

  for (int k0 = 0; k0 < D_; k0 += 32) {
    if (wave == 0) {
      // 2 KB tile = 128 chunks of 16 B; 32 lanes x 4 iterations.
#pragma unroll
      for (int it = 0; it < 4; ++it) {
        const int c  = it * 32 + lane;            // chunk index
        const int r  = c >> 3;                    // row 0..15
        const int cw = (c & 7) * 4;               // word within row
        unsigned long long gaddr =
            (unsigned long long)(xrow + (size_t)r * D_ + k0 + cw);
        unsigned laddr = xt_base + (unsigned)c * 16u;
        asm volatile("global_load_async_to_lds_b128 %0, %1, off"
                     :: "v"(laddr), "v"(gaddr) : "memory");
      }
      asm volatile("s_wait_asynccnt 0x0" ::: "memory");
    }
    __syncthreads();

    // A operand: per lane, two contiguous 8-float chunks -> ds_load_b128 x4.
    v16bf a;
#pragma unroll
    for (int vv = 0; vv < 8; ++vv) {
      int kk = a_k_of(vv, hi);
      a[2 * vv]     = f2bf(xt[m][kk]);
      a[2 * vv + 1] = f2bf(xt[m][kk + 1]);
    }

#pragma unroll
    for (int t = 0; t < 4; ++t) {
      const int n0 = t * 16;
      const __bf16* bp = Wrow + (size_t)(n0 + n) * D_ + k0 + 16 * hi;
      v16bf bm;
#pragma unroll
      for (int e = 0; e < 16; ++e) bm[e] = bp[e];   // contiguous 32 B -> b128 x2
      acc[t] = __builtin_amdgcn_wmma_f32_16x16x32_bf16(
          false, a, false, bm, (short)0, acc[t], false, false);
    }
    __syncthreads();
  }

  if (wave < 2) {
    __bf16* dst = (wave == 0) ? q : k;            // [B][S][H]
#pragma unroll
    for (int t = 0; t < 4; ++t) {
#pragma unroll
      for (int r = 0; r < 8; ++r) {
        int M = r + 8 * hi;
        dst[((size_t)b * S_ + row0 + M) * H_ + t * 16 + n] = f2bf(acc[t][r]);
      }
    }
  } else {
    // vT[B][H][S]
#pragma unroll
    for (int t = 0; t < 4; ++t) {
#pragma unroll
      for (int r = 0; r < 8; ++r) {
        int M = r + 8 * hi;
        vT[((size_t)b * H_ + t * 16 + n) * S_ + row0 + M] = f2bf(acc[t][r]);
      }
    }
  }
}

// ---------------------------------------------------------------------------
// Kernel 2: scores tile = (q 16x64) x (k^T 64x16) * 0.125 -> f32 weights.
// Grid (S/16, S/16, B), block = 32 (one wave per 16x16 tile).
// k stored [S][H] == (k^T)^T, so the B-operand per lane is contiguous.
// ---------------------------------------------------------------------------
__global__ __launch_bounds__(32) void scores_kernel(
    const __bf16* __restrict__ q, const __bf16* __restrict__ k,
    float* __restrict__ w) {
  const int col0 = blockIdx.x * 16;
  const int row0 = blockIdx.y * 16;
  const int b    = blockIdx.z;
  const int lane = threadIdx.x;
  const int m  = lane & 15;
  const int hi = lane >> 4;
  const int n  = lane & 15;

  v8f acc = {};
  const __bf16* qrow = q + ((size_t)b * S_ + row0) * H_;
  const __bf16* krow = k + ((size_t)b * S_ + col0) * H_;

#pragma unroll
  for (int s = 0; s < 2; ++s) {
    const int k0 = s * 32;
    v16bf a, bm;
#pragma unroll
    for (int vv = 0; vv < 8; ++vv) {
      int ka = k0 + a_k_of(vv, hi);
      a[2 * vv]     = qrow[(size_t)m * H_ + ka];
      a[2 * vv + 1] = qrow[(size_t)m * H_ + ka + 1];
    }
    const __bf16* bp = krow + (size_t)n * H_ + k0 + 16 * hi;
#pragma unroll
    for (int e = 0; e < 16; ++e) bm[e] = bp[e];
    acc = __builtin_amdgcn_wmma_f32_16x16x32_bf16(
        false, a, false, bm, (short)0, acc, false, false);
  }

#pragma unroll
  for (int r = 0; r < 8; ++r) {
    int M = r + 8 * hi;
    w[((size_t)b * S_ + row0 + M) * S_ + col0 + n] = acc[r] * 0.125f;  // 1/sqrt(64)
  }
}

// ---------------------------------------------------------------------------
// Kernel 3: row softmax in place over weights (L2-resident). One block/row.
// ---------------------------------------------------------------------------
__global__ __launch_bounds__(256) void softmax_kernel(float* __restrict__ w) {
  __shared__ float sdata[256];
  float* p = w + (size_t)blockIdx.x * S_;
  const int tid = threadIdx.x;

  float vals[S_ / 256];
  float vmax = -3.402823466e+38f;
#pragma unroll
  for (int i = 0; i < S_ / 256; ++i) {
    vals[i] = p[tid + 256 * i];
    vmax = fmaxf(vmax, vals[i]);
  }
  sdata[tid] = vmax;
  __syncthreads();
  for (int s = 128; s > 0; s >>= 1) {
    if (tid < s) sdata[tid] = fmaxf(sdata[tid], sdata[tid + s]);
    __syncthreads();
  }
  vmax = sdata[0];
  __syncthreads();

  float vsum = 0.0f;
#pragma unroll
  for (int i = 0; i < S_ / 256; ++i) {
    vals[i] = __expf(vals[i] - vmax);
    vsum += vals[i];
  }
  sdata[tid] = vsum;
  __syncthreads();
  for (int s = 128; s > 0; s >>= 1) {
    if (tid < s) sdata[tid] += sdata[tid + s];
    __syncthreads();
  }
  const float inv = 1.0f / sdata[0];
#pragma unroll
  for (int i = 0; i < S_ / 256; ++i) p[tid + 256 * i] = vals[i] * inv;
}

// ---------------------------------------------------------------------------
// Kernel 4: out = weights @ v.  Grid (S/16, B), block = 32 (one wave).
// K = S = 2048 -> 64 WMMA steps; 4 accumulators over H=64.
// A = weights f32 (L2-resident) converted on the fly; B from vT (contiguous).
// ---------------------------------------------------------------------------
__global__ __launch_bounds__(32) void out_kernel(
    const float* __restrict__ w, const __bf16* __restrict__ vT,
    float* __restrict__ out) {
  const int row0 = blockIdx.x * 16;
  const int b    = blockIdx.y;
  const int lane = threadIdx.x;
  const int m  = lane & 15;
  const int hi = lane >> 4;
  const int n  = lane & 15;

  v8f acc[4] = {};
  const float*  wrow   = w + ((size_t)b * S_ + row0) * S_;
  const __bf16* vTbase = vT + (size_t)b * H_ * S_;   // [H][S]

  for (int k0 = 0; k0 < S_; k0 += 32) {
    v16bf a;
#pragma unroll
    for (int vv = 0; vv < 8; ++vv) {
      int ka = k0 + a_k_of(vv, hi);
      a[2 * vv]     = f2bf(wrow[(size_t)m * S_ + ka]);
      a[2 * vv + 1] = f2bf(wrow[(size_t)m * S_ + ka + 1]);
    }
#pragma unroll
    for (int t = 0; t < 4; ++t) {
      const __bf16* bp = vTbase + (size_t)(t * 16 + n) * S_ + k0 + 16 * hi;
      v16bf bm;
#pragma unroll
      for (int e = 0; e < 16; ++e) bm[e] = bp[e];   // contiguous 32 B
      acc[t] = __builtin_amdgcn_wmma_f32_16x16x32_bf16(
          false, a, false, bm, (short)0, acc[t], false, false);
    }
  }

#pragma unroll
  for (int t = 0; t < 4; ++t) {
#pragma unroll
    for (int r = 0; r < 8; ++r) {
      int M = r + 8 * hi;
      out[((size_t)b * S_ + row0 + M) * H_ + t * 16 + n] = acc[t][r];
    }
  }
}

// ---------------------------------------------------------------------------
extern "C" void kernel_launch(void* const* d_in, const int* in_sizes, int n_in,
                              void* d_out, int out_size, void* d_ws,
                              size_t ws_size, hipStream_t stream) {
  const float* x  = (const float*)d_in[0];
  const float* Wq = (const float*)d_in[1];
  const float* Wk = (const float*)d_in[2];
  const float* Wv = (const float*)d_in[3];

  float* out     = (float*)d_out;                  // [B,S,H]
  float* weights = out + (size_t)B_ * S_ * H_;     // [B,S,S]

  __bf16* q  = (__bf16*)d_ws;                      // [B][S][H]  2 MB
  __bf16* k  = q + (size_t)B_ * S_ * H_;           // [B][S][H]  2 MB
  __bf16* vT = k + (size_t)B_ * S_ * H_;           // [B][H][S]  2 MB
  __bf16* WT = vT + (size_t)B_ * S_ * H_;          // [3][H][D]  384 KB

  prep_kernel<<<dim3(H_, 3), 256, 0, stream>>>(Wq, Wk, Wv, WT);
  proj_kernel<<<dim3(S_ / 16, B_), 96, 0, stream>>>(x, WT, q, k, vT);
  scores_kernel<<<dim3(S_ / 16, S_ / 16, B_), 32, 0, stream>>>(q, k, weights);
  softmax_kernel<<<dim3(B_ * S_), 256, 0, stream>>>(weights);
  out_kernel<<<dim3(S_ / 16, B_), 32, 0, stream>>>(weights, vT, out);
}
